// LesionTriplet_52536039964792
// MI455X (gfx1250) — matrix operand order repaired and checked by reading, not compiled, not run on hardware
//
#include <hip/hip_runtime.h>
#include <hip/hip_bf16.h>
#include <math.h>

typedef __bf16 v16bf __attribute__((ext_vector_type(16)));
typedef float  v8f   __attribute__((ext_vector_type(8)));
typedef float  vf4   __attribute__((ext_vector_type(4)));

#define N_ROWS   8192
#define DIM      128
#define B_BATCH  256
#define INV_TEMP (1.0f / 0.07f)
#define NEG_BIG  -1e30f

// ---------------------------------------------------------------------------
// Kernel 0: zero the small accumulators (graph-capture safe, no memset).
// ---------------------------------------------------------------------------
__global__ void LT_zero_kernel(float* gsum /* 2*DIM + 2 floats */, float* out) {
    int t = threadIdx.x;
    if (t < 2 * DIM + 2) gsum[t] = 0.0f;
    if (t == 0) *out = 0.0f;
}

// ---------------------------------------------------------------------------
// Kernel 1: f32 -> bf16 convert, valid mask (additive form), labels -> int32.
// One block (128 threads) per row.
// ---------------------------------------------------------------------------
__global__ void LT_prep_kernel(const float* __restrict__ f,
                               const int* __restrict__ lab,
                               __hip_bfloat16* __restrict__ fb,
                               float* __restrict__ maskadd,
                               int* __restrict__ labi) {
    int r = blockIdx.x;
    int k = threadIdx.x;
    __shared__ int nz;
    if (k == 0) nz = 0;
    __syncthreads();
    float x = f[r * DIM + k];
    fb[r * DIM + k] = __float2bfloat16(x);
    if (x != 0.0f) nz = 1;  // benign race: any writer sets 1
    __syncthreads();
    if (k == 0) {
        maskadd[r] = nz ? 0.0f : NEG_BIG;
        labi[r] = lab[r];
    }
}

// ---------------------------------------------------------------------------
// Kernel 2: per-class feature sums g_c[k] = sum_{rows with label c} f[r][k]
// (invalid rows are all-zero, contribute nothing), and valid per-class counts.
// ---------------------------------------------------------------------------
__global__ void LT_classsum_kernel(const float* __restrict__ f,
                                   const int* __restrict__ labi,
                                   const float* __restrict__ maskadd,
                                   float* __restrict__ g /* [2][DIM] */,
                                   float* __restrict__ cntf /* [2] */) {
    const int rows_per_blk = N_ROWS / 64;  // grid 64
    int r0 = blockIdx.x * rows_per_blk;
    int tid = threadIdx.x;
    if (tid < DIM) {
        float a0 = 0.0f, a1 = 0.0f;
        for (int r = r0; r < r0 + rows_per_blk; ++r) {
            float x = f[r * DIM + tid];
            if (labi[r] == 0) a0 += x; else a1 += x;
        }
        atomicAdd(&g[0 * DIM + tid], a0);
        atomicAdd(&g[1 * DIM + tid], a1);
    } else if (tid < DIM + 2) {
        int c = tid - DIM;
        float cnt = 0.0f;
        for (int r = r0; r < r0 + rows_per_blk; ++r)
            if (labi[r] == c && maskadd[r] == 0.0f) cnt += 1.0f;
        atomicAdd(&cntf[c], cnt);
    }
}

// ---------------------------------------------------------------------------
// Kernel 3: fused WMMA GEMM (S = f f^T / T) + streaming logsumexp + loss.
// 512 blocks of 256 threads (8 waves). Block owns 16 anchor rows; A-matrix
// fragments (16x128 bf16, 4 K-chunks of 32) stay resident in VGPRs.
// Wave w handles column tile jb = it*128 + w*16 each iteration.
// ---------------------------------------------------------------------------
__global__ __launch_bounds__(256) void LT_lse_kernel(
    const __hip_bfloat16* __restrict__ fb,
    const float* __restrict__ f,
    const float* __restrict__ maskadd,
    const int* __restrict__ labi,
    const float* __restrict__ g,
    const float* __restrict__ cntf,
    float* __restrict__ out) {
    const int wg   = blockIdx.x;
    const int tid  = threadIdx.x;
    const int wave = tid >> 5;
    const int lane = tid & 31;
    const int half = lane >> 4;   // K-half selector per ISA VGPR layout
    const int mn   = lane & 15;   // A: row m of tile; B/C: column n of tile

    union Frag { v16bf v; vf4 f4[2]; };

    // ---- A fragments: anchor row (wg*16 + mn), 4 chunks of K=32 ----
    // 16-bit A 16x32 layout: lanes 0-15 hold K=(chunk*32 + 8*half)+0..7 in
    // VGPRs 0-3 and K+16 in VGPRs 4-7 (pairs packed per dword).
    const __hip_bfloat16* arow = fb + (size_t)(wg * 16 + mn) * DIM;
    Frag A[4];
#pragma unroll
    for (int c = 0; c < 4; ++c) {
        A[c].f4[0] = *(const vf4*)(arow + 32 * c + 8 * half);
        A[c].f4[1] = *(const vf4*)(arow + 32 * c + 16 + 8 * half);
    }

    // Online-softmax state: 8 rows per lane (m = v + 8*half in C/D layout).
    float mx[8], sm[8];
#pragma unroll
    for (int v = 0; v < 8; ++v) { mx[v] = -3.0e38f; sm[v] = 0.0f; }

    for (int it = 0; it < N_ROWS / 128; ++it) {
        const int col = it * 128 + wave * 16 + mn;  // this lane's column
        const __hip_bfloat16* brow = fb + (size_t)col * DIM;

        if (it + 1 < N_ROWS / 128)  // CDNA5 global_prefetch_b8 for next tile
            __builtin_prefetch((const void*)(brow + 128 * DIM), 0, 1);

        // B 32x16 bf16 layout: lane n=mn, VGPRs 0-3 hold K = chunk*32 +
        // 8*half + 0..7 and VGPRs 4-7 hold K+16 (same order as A).
        v8f acc = {};
#pragma unroll
        for (int c = 0; c < 4; ++c) {
            Frag B;
            B.f4[0] = *(const vf4*)(brow + 32 * c + 8 * half);
            B.f4[1] = *(const vf4*)(brow + 32 * c + 16 + 8 * half);
            acc = __builtin_amdgcn_wmma_f32_16x16x32_bf16(
                false, A[c].v, false, B.v, (short)0, acc, false, false);
        }

        const float cmask = maskadd[col];  // 0 (valid col) or -1e30
#pragma unroll
        for (int v = 0; v < 8; ++v) {
            float s  = acc[v] * INV_TEMP + cmask;
            float m2 = fmaxf(mx[v], s);
            sm[v] = sm[v] * __expf(mx[v] - m2) + __expf(s - m2);
            mx[v] = m2;
        }
    }

    // ---- combine (max,sum) across lanes and waves through LDS ----
    __shared__ float redm[8][32][8];
    __shared__ float reds[8][32][8];
#pragma unroll
    for (int v = 0; v < 8; ++v) {
        redm[wave][lane][v] = mx[v];
        reds[wave][lane][v] = sm[v];
    }
    __syncthreads();

    if (tid < 16) {
        const int m = tid;            // anchor row within tile
        const int h = m >> 3;         // which lane-half holds this m
        const int v = m & 7;          // which accumulator VGPR
        float M = -3.0e38f, S = 0.0f;
        for (int w = 0; w < 8; ++w) {
            for (int l = 16 * h; l < 16 * h + 16; ++l) {
                float m1 = redm[w][l][v], s1 = reds[w][l][v];
                float M2 = fmaxf(M, m1);
                S = S * __expf(M - M2) + s1 * __expf(m1 - M2);
                M = M2;
            }
        }
        const float lse = M + __logf(S);

        const int r = wg * 16 + m;
        if (maskadd[r] == 0.0f) {     // anchor is valid
            const int lb = labi[r];
            const float* gc = g + lb * DIM;
            const float* fr = f + (size_t)r * DIM;
            float dot = 0.0f;
#pragma unroll 8
            for (int k = 0; k < DIM; ++k) dot += fr[k] * gc[k];
            const float possum = dot * INV_TEMP;
            const float per = (possum - cntf[lb] * lse) / (-(float)DIM);
            atomicAdd(out, per / (float)B_BATCH);
        }
    }
}

// ---------------------------------------------------------------------------
extern "C" void kernel_launch(void* const* d_in, const int* in_sizes, int n_in,
                              void* d_out, int out_size, void* d_ws, size_t ws_size,
                              hipStream_t stream) {
    const float* feats = (const float*)d_in[0];
    const int*   lab   = (const int*)d_in[1];

    char* ws = (char*)d_ws;
    __hip_bfloat16* fb      = (__hip_bfloat16*)ws;                         // 2 MB
    float*          maskadd = (float*)(ws + (size_t)N_ROWS * DIM * 2);     // 32 KB
    int*            labi    = (int*)  (ws + (size_t)N_ROWS * DIM * 2 + N_ROWS * 4);
    float*          g       = (float*)(ws + (size_t)N_ROWS * DIM * 2 + N_ROWS * 8);
    float*          cntf    = g + 2 * DIM;                                 // contiguous

    float* out = (float*)d_out;

    LT_zero_kernel<<<1, 320, 0, stream>>>(g, out);
    LT_prep_kernel<<<N_ROWS, DIM, 0, stream>>>(feats, lab, fb, maskadd, labi);
    LT_classsum_kernel<<<64, 256, 0, stream>>>(feats, labi, maskadd, g, cntf);
    LT_lse_kernel<<<N_ROWS / 16, 256, 0, stream>>>(fb, feats, maskadd, labi, g,
                                                   cntf, out);
}
